// MultiHeadAttention_52235392254539
// MI455X (gfx1250) — matrix operand order repaired and checked by reading, not compiled, not run on hardware
//
#include <hip/hip_runtime.h>
#include <hip/hip_bf16.h>
#include <math.h>

// ---------------------------------------------------------------------------
// Types for CDNA5 WMMA (wave32): v_wmma_f32_16x16x32_bf16
// ---------------------------------------------------------------------------
typedef __attribute__((ext_vector_type(8)))  float  v8f;
typedef __attribute__((ext_vector_type(8)))  __bf16 bf16x8;
typedef __attribute__((ext_vector_type(16))) __bf16 v16bf;
typedef __bf16 bf16;

union BF16Frag { bf16x8 h[2]; v16bf v; };

__device__ __forceinline__ v8f vzero8() {
    v8f z;
#pragma unroll
    for (int i = 0; i < 8; ++i) z[i] = 0.0f;
    return z;
}

__device__ __forceinline__ v8f wmma_bf16(const v16bf& a, const v16bf& b, const v8f& c) {
    // (neg_a, A, neg_b, B, c_mod, C, reuse_a, reuse_b)
    return __builtin_amdgcn_wmma_f32_16x16x32_bf16(false, a, false, b, (short)0, c, false, false);
}

__device__ __forceinline__ float rmax16(float v) {
    v = fmaxf(v, __shfl_xor(v, 1));
    v = fmaxf(v, __shfl_xor(v, 2));
    v = fmaxf(v, __shfl_xor(v, 4));
    v = fmaxf(v, __shfl_xor(v, 8));
    return v;
}
__device__ __forceinline__ float rsum16(float v) {
    v += __shfl_xor(v, 1);
    v += __shfl_xor(v, 2);
    v += __shfl_xor(v, 4);
    v += __shfl_xor(v, 8);
    return v;
}

// ---------------------------------------------------------------------------
// GEMM: C[M,N] = A[M,K] @ W[K,N] + bias   (fp32 in/out, bf16 WMMA inside)
// Tile 128x128x32, 256 threads = 8 waves (4 along M, 2 along N).
// Software-pipelined: global loads for tile k+1 are issued before the
// fragment loads + 8 WMMAs of tile k, hiding HBM latency behind the matrix
// pipe (s_wait_loadcnt only lands at the next iteration's LDS store).
// ---------------------------------------------------------------------------
#define GLDA 40   // LDS row stride (bf16 elems) for As  [128][40]
#define GLDB 40   // LDS row stride for Bs (stored [n][k]) [128][40]

__global__ __launch_bounds__(256)
void gemm_bias_wmma(const float* __restrict__ A, const float* __restrict__ W,
                    const float* __restrict__ bias, float* __restrict__ C,
                    int M, int N, int K) {
    __shared__ bf16 As[128 * GLDA];
    __shared__ bf16 Bs[128 * GLDB];

    const int tid  = threadIdx.x;
    const int lane = tid & 31;
    const int wid  = tid >> 5;
    const int wm   = wid & 3;     // wave row (4 x 32 rows)
    const int wn   = wid >> 2;    // wave col (2 x 64 cols)
    const int m0   = blockIdx.y * 128;
    const int n0   = blockIdx.x * 128;

    const int hlf = lane >> 4;    // which 16-lane half
    const int l16 = lane & 15;

    v8f acc[2][4];
#pragma unroll
    for (int i = 0; i < 2; ++i)
#pragma unroll
        for (int j = 0; j < 4; ++j) acc[i][j] = vzero8();

    const int arow  = tid >> 3;   // 0..31 (+32 per pass)
    const int acol4 = tid & 7;    // float4 within 32 cols
    const int bk    = tid >> 5;   // 0..7  (+8 per pass)
    const int bn4   = tid & 31;   // float4 group along N (128 cols)

    float4 aR[4], bR[4];
    // ---- prologue: load tile k=0 into registers ----
#pragma unroll
    for (int p = 0; p < 4; ++p) {
        aR[p] = *(const float4*)(A + (size_t)(m0 + arow + p * 32) * K + acol4 * 4);
        bR[p] = *(const float4*)(W + (size_t)(bk + p * 8) * N + n0 + bn4 * 4);
    }

    for (int k0 = 0; k0 < K; k0 += 32) {
        // ---- commit staged registers to LDS (bf16 convert; B transposed) ----
#pragma unroll
        for (int p = 0; p < 4; ++p) {
            const int r = arow + p * 32;
            bf16* dst = &As[r * GLDA + acol4 * 4];
            dst[0] = (bf16)aR[p].x; dst[1] = (bf16)aR[p].y;
            dst[2] = (bf16)aR[p].z; dst[3] = (bf16)aR[p].w;
            const int kk = bk + p * 8;
            Bs[(bn4 * 4 + 0) * GLDB + kk] = (bf16)bR[p].x;
            Bs[(bn4 * 4 + 1) * GLDB + kk] = (bf16)bR[p].y;
            Bs[(bn4 * 4 + 2) * GLDB + kk] = (bf16)bR[p].z;
            Bs[(bn4 * 4 + 3) * GLDB + kk] = (bf16)bR[p].w;
        }
        __syncthreads();

        // ---- issue global loads for tile k+1 (overlap with WMMA below) ----
        const int kn = k0 + 32;
        if (kn < K) {
#pragma unroll
            for (int p = 0; p < 4; ++p) {
                aR[p] = *(const float4*)(A + (size_t)(m0 + arow + p * 32) * K + kn + acol4 * 4);
                bR[p] = *(const float4*)(W + (size_t)(kn + bk + p * 8) * N + n0 + bn4 * 4);
            }
        }

        // ---- fragments (ISA lane layout, 16-bit A 16x32 / B 32x16) ----
        BF16Frag afrag[2], bfrag[4];
#pragma unroll
        for (int i = 0; i < 2; ++i) {
            const int m  = wm * 32 + i * 16 + l16;
            const int kb = hlf ? 8 : 0;            // lanes 0-15: K 0-7,16-23 ; 16-31: K 8-15,24-31
            afrag[i].h[0] = *(const bf16x8*)&As[m * GLDA + kb];
            afrag[i].h[1] = *(const bf16x8*)&As[m * GLDA + kb + 16];
        }
#pragma unroll
        for (int j = 0; j < 4; ++j) {
            const int n  = wn * 64 + j * 16 + l16;
            const int kb = hlf ? 8 : 0;            // lanes 0-15: K 0-7,16-23 ; 16-31: K 8-15,24-31
            bfrag[j].h[0] = *(const bf16x8*)&Bs[n * GLDB + kb];
            bfrag[j].h[1] = *(const bf16x8*)&Bs[n * GLDB + kb + 16];
        }
#pragma unroll
        for (int i = 0; i < 2; ++i)
#pragma unroll
            for (int j = 0; j < 4; ++j)
                acc[i][j] = wmma_bf16(afrag[i].v, bfrag[j].v, acc[i][j]);
        __syncthreads();
    }

    // ---- epilogue: C 16x16 layout => row = r + 8*hlf, col = l16 ----
#pragma unroll
    for (int i = 0; i < 2; ++i)
#pragma unroll
        for (int j = 0; j < 4; ++j) {
            const int col = n0 + wn * 64 + j * 16 + l16;
            const float bi = bias[col];
#pragma unroll
            for (int r = 0; r < 8; ++r) {
                const int row = m0 + wm * 32 + i * 16 + r + hlf * 8;
                C[(size_t)row * N + col] = acc[i][j][r] + bi;
            }
        }
}

// ---------------------------------------------------------------------------
// RoPE (interleaved pairs), in-place on [rows, cols] where col = head*128 + d
// ---------------------------------------------------------------------------
__global__ void rope_kernel(float* __restrict__ x, int rows, int cols, int T) {
    const int pairs = cols >> 1;
    const int idx = blockIdx.x * blockDim.x + threadIdx.x;
    if (idx >= rows * pairs) return;
    const int row = idx / pairs;
    const int pc  = idx - row * pairs;
    const int t   = row % T;              // rows = B*T
    const int p   = pc & 63;              // pair index within a head (D/2 = 64)
    const float invf = __expf(-((float)(2 * p) / 128.0f) * 9.210340371976184f); // ln(10000)
    const float ang  = (float)t * invf;
    const float c = __cosf(ang), s = __sinf(ang);
    float* px = x + (size_t)row * cols + pc * 2;
    const float x0 = px[0], x1 = px[1];
    px[0] = x0 * c - x1 * s;
    px[1] = x1 * c + x0 * s;
}

// ---------------------------------------------------------------------------
// Flash attention (causal, GQA): per block = 64 query rows of one query head.
// 128 threads = 4 waves, each wave owns 16 query rows. Key blocks of 32.
// QK^T and P@V both on v_wmma_f32_16x16x32_bf16; online softmax in f32.
// Next K/V block is prefetched (global_prefetch_b8) while current block's
// WMMAs + softmax execute.
// q: [B*T, 2048] (h*128+d), k/v: [B*T, 512] (kvh*128+d), out: [B*T, 2048]
// ---------------------------------------------------------------------------
#define QLD 136
#define KLD 136
#define VLD 40
#define PLD 40
#define ATT_SCALE 0.08838834764831845f

__global__ __launch_bounds__(128)
void flash_attn_wmma(const float* __restrict__ q, const float* __restrict__ k,
                     const float* __restrict__ v, float* __restrict__ o) {
    __shared__ bf16 Qs[64 * QLD];        // [qrow][d]
    __shared__ bf16 Ks[32 * KLD];        // [key][d]
    __shared__ bf16 Vs[128 * VLD];       // [d][key]  (transposed)
    __shared__ bf16 Ps[4 * 16 * PLD];    // per-wave P staging [16][keys]

    const int tid  = threadIdx.x;
    const int lane = tid & 31;
    const int w    = tid >> 5;           // wave 0..3
    const int hlf  = lane >> 4;
    const int l16  = lane & 15;

    const int head = blockIdx.y;         // 0..B*H-1
    const int b    = head >> 4;
    const int h    = head & 15;
    const int kvh  = h >> 2;             // G = 4
    const int r0   = blockIdx.x * 64;

    const size_t qbase  = ((size_t)b * 2048 + r0) * 2048 + h * 128;
    const size_t kvbase = ((size_t)b * 2048) * 512 + kvh * 128;

    // ---- stage Q tile: 64 x 128 fp32 -> bf16 ----
#pragma unroll
    for (int it = 0; it < 16; ++it) {
        const int idx = tid + it * 128;          // 0..2047 float4s
        const int row = idx >> 5;
        const int c4  = idx & 31;
        const float4 f = *(const float4*)(q + qbase + (size_t)row * 2048 + c4 * 4);
        bf16* dst = &Qs[row * QLD + c4 * 4];
        dst[0] = (bf16)f.x; dst[1] = (bf16)f.y; dst[2] = (bf16)f.z; dst[3] = (bf16)f.w;
    }

    float mrow[8], lrow[8];
    v8f oacc[8];
#pragma unroll
    for (int r = 0; r < 8; ++r) { mrow[r] = -1e30f; lrow[r] = 0.0f; }
#pragma unroll
    for (int d = 0; d < 8; ++d) oacc[d] = vzero8();

    const int qlocal = w * 16;
    const int nblocks = (r0 + 64) >> 5;          // causal: keys up to r0+63

    for (int jb = 0; jb < nblocks; ++jb) {
        const int j0 = jb * 32;
        __syncthreads();                         // protect Ks/Vs reuse
        // ---- stage K (32x128 -> [key][d]) and V (32x128 -> [d][key]) ----
#pragma unroll
        for (int it = 0; it < 8; ++it) {
            const int idx = tid + it * 128;      // 0..1023 float4s
            const int row = idx >> 5;
            const int c4  = idx & 31;
            const float4 f = *(const float4*)(k + kvbase + (size_t)(j0 + row) * 512 + c4 * 4);
            bf16* dst = &Ks[row * KLD + c4 * 4];
            dst[0] = (bf16)f.x; dst[1] = (bf16)f.y; dst[2] = (bf16)f.z; dst[3] = (bf16)f.w;
            const float4 g = *(const float4*)(v + kvbase + (size_t)(j0 + row) * 512 + c4 * 4);
            Vs[(c4 * 4 + 0) * VLD + row] = (bf16)g.x;
            Vs[(c4 * 4 + 1) * VLD + row] = (bf16)g.y;
            Vs[(c4 * 4 + 2) * VLD + row] = (bf16)g.z;
            Vs[(c4 * 4 + 3) * VLD + row] = (bf16)g.w;
        }
        __syncthreads();

        // ---- prefetch next K/V block (gfx1250 global_prefetch_b8) ----
        if (jb + 1 < nblocks) {
            const size_t nb0 = kvbase + (size_t)(j0 + 32 + (tid >> 2)) * 512 + (tid & 3) * 32;
            __builtin_prefetch(k + nb0, 0, 1);
            __builtin_prefetch(v + nb0, 0, 1);
        }

        // ---- S = Q K^T : 2 key sub-blocks x 4 k-steps over d ----
        v8f s[2];
        s[0] = vzero8(); s[1] = vzero8();
#pragma unroll
        for (int kk = 0; kk < 128; kk += 32) {
            BF16Frag a;
            const int m  = qlocal + l16;
            const int kb = hlf ? 8 : 0;
            a.h[0] = *(const bf16x8*)&Qs[m * QLD + kk + kb];
            a.h[1] = *(const bf16x8*)&Qs[m * QLD + kk + kb + 16];
#pragma unroll
            for (int nb = 0; nb < 2; ++nb) {
                BF16Frag bb;
                const int n   = nb * 16 + l16;
                const int kb2 = hlf ? 8 : 0;
                bb.h[0] = *(const bf16x8*)&Ks[n * KLD + kk + kb2];
                bb.h[1] = *(const bf16x8*)&Ks[n * KLD + kk + kb2 + 16];
                s[nb] = wmma_bf16(a.v, bb.v, s[nb]);
            }
        }

        // ---- scale + causal mask (C layout: row = r + 8*hlf, col = l16) ----
#pragma unroll
        for (int nb = 0; nb < 2; ++nb)
#pragma unroll
            for (int r = 0; r < 8; ++r) {
                const int qr = r0 + qlocal + r + hlf * 8;
                const int kc = j0 + nb * 16 + l16;
                s[nb][r] = (kc <= qr) ? s[nb][r] * ATT_SCALE : -1e30f;
            }

        // ---- online softmax ----
        float alpha[8];
#pragma unroll
        for (int r = 0; r < 8; ++r) {
            float mx = rmax16(fmaxf(s[0][r], s[1][r]));
            const float mnew = fmaxf(mrow[r], mx);
            alpha[r] = __expf(mrow[r] - mnew);
            mrow[r] = mnew;
            s[0][r] = __expf(s[0][r] - mnew);
            s[1][r] = __expf(s[1][r] - mnew);
            const float rs = rsum16(s[0][r] + s[1][r]);
            lrow[r] = lrow[r] * alpha[r] + rs;
        }

        // ---- stage P (C-layout -> LDS -> A-layout); per-wave region, LDS is
        //      in-order within a wave so no barrier needed ----
        bf16* pw = &Ps[w * 16 * PLD];
#pragma unroll
        for (int nb = 0; nb < 2; ++nb)
#pragma unroll
            for (int r = 0; r < 8; ++r)
                pw[(r + hlf * 8) * PLD + nb * 16 + l16] = (bf16)s[nb][r];

        // ---- rescale O, accumulate O += P @ V ----
#pragma unroll
        for (int d = 0; d < 8; ++d)
#pragma unroll
            for (int r = 0; r < 8; ++r) oacc[d][r] *= alpha[r];

        BF16Frag pa;
        {
            const int kb = hlf ? 8 : 0;
            pa.h[0] = *(const bf16x8*)&pw[l16 * PLD + kb];
            pa.h[1] = *(const bf16x8*)&pw[l16 * PLD + kb + 16];
        }
#pragma unroll
        for (int db = 0; db < 8; ++db) {
            BF16Frag bb;
            const int n   = db * 16 + l16;
            const int kb2 = hlf ? 8 : 0;
            bb.h[0] = *(const bf16x8*)&Vs[n * VLD + kb2];
            bb.h[1] = *(const bf16x8*)&Vs[n * VLD + kb2 + 16];
            oacc[db] = wmma_bf16(pa.v, bb.v, oacc[db]);
        }
    }

    // ---- normalize + write out to [B*T, 2048] at col h*128+d ----
    float inv[8];
#pragma unroll
    for (int r = 0; r < 8; ++r) inv[r] = 1.0f / lrow[r];
#pragma unroll
    for (int db = 0; db < 8; ++db)
#pragma unroll
        for (int r = 0; r < 8; ++r) {
            const int qr = r0 + qlocal + r + hlf * 8;
            o[((size_t)b * 2048 + qr) * 2048 + h * 128 + db * 16 + l16] = oacc[db][r] * inv[r];
        }
}

// ---------------------------------------------------------------------------
// Launch
// ---------------------------------------------------------------------------
extern "C" void kernel_launch(void* const* d_in, const int* in_sizes, int n_in,
                              void* d_out, int out_size, void* d_ws, size_t ws_size,
                              hipStream_t stream) {
    (void)in_sizes; (void)n_in; (void)out_size; (void)ws_size;
    const float* x  = (const float*)d_in[0];
    const float* Wq = (const float*)d_in[1];
    const float* bq = (const float*)d_in[2];
    const float* Wk = (const float*)d_in[3];
    const float* bk = (const float*)d_in[4];
    const float* Wv = (const float*)d_in[5];
    const float* bv = (const float*)d_in[6];
    const float* Wo = (const float*)d_in[7];
    const float* bo = (const float*)d_in[8];
    float* out = (float*)d_out;

    const int M = 4096;   // B*T
    const int C = 2048;
    const int KV = 512;   // KVH*D

    float* qb  = (float*)d_ws;                    // [4096, 2048]
    float* kb  = qb + (size_t)M * C;              // [4096, 512]
    float* vb  = kb + (size_t)M * KV;             // [4096, 512]
    float* att = vb + (size_t)M * KV;             // [4096, 2048]

    // Projections
    gemm_bias_wmma<<<dim3(C / 128, M / 128), 256, 0, stream>>>(x, Wq, bq, qb, M, C, C);
    gemm_bias_wmma<<<dim3(KV / 128, M / 128), 256, 0, stream>>>(x, Wk, bk, kb, M, KV, C);
    gemm_bias_wmma<<<dim3(KV / 128, M / 128), 256, 0, stream>>>(x, Wv, bv, vb, M, KV, C);

    // RoPE on Q and K
    {
        const int nq = M * (C / 2);
        rope_kernel<<<(nq + 255) / 256, 256, 0, stream>>>(qb, M, C, 2048);
        const int nk = M * (KV / 2);
        rope_kernel<<<(nk + 255) / 256, 256, 0, stream>>>(kb, M, KV, 2048);
    }

    // Flash attention: grid = (T/64 query tiles, B*H heads)
    flash_attn_wmma<<<dim3(2048 / 64, 32), 128, 0, stream>>>(qb, kb, vb, att);

    // Output projection
    gemm_bias_wmma<<<dim3(C / 128, M / 128), 256, 0, stream>>>(att, Wo, bo, out, M, C, C);
}
